// MoE_hard_61040075210967
// MI455X (gfx1250) — matrix operand order repaired and checked by reading, not compiled, not run on hardware
//
#include <hip/hip_runtime.h>
#include <hip/hip_fp16.h>

typedef __attribute__((ext_vector_type(16))) _Float16 v16h;
typedef __attribute__((ext_vector_type(8)))  float    v8f;
typedef __attribute__((ext_vector_type(4)))  float    v4f;

#define BB 256
#define NN 64
#define DD 1024
#define EE 8
#define HH 1024
#define TOKENS (BB*NN)          /* 16384 */
#define SLOTS_PER_E 256
#define NSLOTS (EE*SLOTS_PER_E) /* 2048 */

// pack two f32 -> one dword of two f16 (little-endian: lo half first)
__device__ __forceinline__ unsigned int pk2(float lo, float hi) {
    union { _Float16 h[2]; unsigned int u; } q;
    q.h[0] = (_Float16)lo; q.h[1] = (_Float16)hi;
    return q.u;
}

// ---------------------------------------------------------------------------
// Kernel 1: gating. One wave32 per token.
//   gate_scores = renorm(top2(softmax(x@gate_W + gate_b)))   -> d_out tail
//   local_scores = x@lg_W + lg_b                              -> workspace
// ---------------------------------------------------------------------------
__global__ __launch_bounds__(256) void k_gate(
    const float* __restrict__ x,
    const float* __restrict__ gW, const float* __restrict__ gB,
    const float* __restrict__ lgW, const float* __restrict__ lgB,
    float* __restrict__ gateOut, float* __restrict__ localScores)
{
    const int wave = threadIdx.x >> 5;
    const int lane = threadIdx.x & 31;
    const int tok  = blockIdx.x * 8 + wave;
    const float* xr = x + (size_t)tok * DD;

    float acc[EE];
#pragma unroll
    for (int e = 0; e < EE; ++e) acc[e] = 0.f;
    float accL = 0.f;

    for (int d = lane; d < DD; d += 32) {
        const float xv = xr[d];
        const float* gwr = gW + (size_t)d * EE;   // 8 consecutive floats
#pragma unroll
        for (int e = 0; e < EE; ++e) acc[e] += xv * gwr[e];
        accL += xv * lgW[d];
    }
    // wave32 reduction
#pragma unroll
    for (int off = 16; off > 0; off >>= 1) {
#pragma unroll
        for (int e = 0; e < EE; ++e) acc[e] += __shfl_xor(acc[e], off, 32);
        accL += __shfl_xor(accL, off, 32);
    }

    if (lane == 0) {
        float s[EE];
        float mx = -1e30f;
#pragma unroll
        for (int e = 0; e < EE; ++e) { s[e] = acc[e] + gB[e]; mx = s[e] > mx ? s[e] : mx; }
        float sum = 0.f;
#pragma unroll
        for (int e = 0; e < EE; ++e) { s[e] = __expf(s[e] - mx); sum += s[e]; }
        const float inv = 1.f / sum;
#pragma unroll
        for (int e = 0; e < EE; ++e) s[e] *= inv;
        // top-2 (strict > keeps earliest index on ties, like lax.top_k)
        int i1 = 0;
#pragma unroll
        for (int e = 1; e < EE; ++e) if (s[e] > s[i1]) i1 = e;
        int i2 = -1;
#pragma unroll
        for (int e = 0; e < EE; ++e)
            if (e != i1 && (i2 < 0 || s[e] > s[i2])) i2 = e;
        const float denom = s[i1] + s[i2] + 1e-9f;
        float* go = gateOut + (size_t)tok * EE;
#pragma unroll
        for (int e = 0; e < EE; ++e)
            go[e] = (e == i1 || e == i2) ? s[e] / denom : 0.f;

        localScores[tok] = accL + lgB[0];
    }
}

// ---------------------------------------------------------------------------
// Kernel 2: per-batch local argmax + build per-expert slot lists.
// Single block of 256 threads (thread == batch index). Slot order from
// atomicAdd is non-deterministic, but every derived quantity (counts, the
// per-row 2-way float accumulation) is order-invariant -> deterministic out.
// ---------------------------------------------------------------------------
__global__ __launch_bounds__(256) void k_dispatch(
    const float* __restrict__ localScores, const float* __restrict__ gateOut,
    int* __restrict__ counts, int* __restrict__ slotRow, float* __restrict__ slotGate)
{
    const int t = threadIdx.x;
    if (t < EE) counts[t] = 0;
    for (int i = t; i < NSLOTS; i += 256) { slotRow[i] = -1; slotGate[i] = 0.f; }
    __syncthreads();

    const float* ls = localScores + (size_t)t * NN;
    int best = 0; float bv = ls[0];
    for (int n = 1; n < NN; ++n) { float v = ls[n]; if (v > bv) { bv = v; best = n; } }
    const int row = t * NN + best;

    const float* g = gateOut + (size_t)row * EE;
#pragma unroll
    for (int e = 0; e < EE; ++e) {
        const float ge = g[e];
        if (ge > 0.f) {
            const int slot = atomicAdd(&counts[e], 1);
            slotRow [e * SLOTS_PER_E + slot] = row;
            slotGate[e * SLOTS_PER_E + slot] = ge;
        }
    }
}

// ---------------------------------------------------------------------------
// Kernel 3: zero the [B,N,H] output region (float4 stores).
// ---------------------------------------------------------------------------
__global__ __launch_bounds__(256) void k_zero(v4f* __restrict__ p)
{
    const size_t i = (size_t)blockIdx.x * 256 + threadIdx.x;
    v4f z = {0.f, 0.f, 0.f, 0.f};
    p[i] = z;
}

// ---------------------------------------------------------------------------
// LDS fragment layouts (per CDNA5 ISA 16-bit A / B tables), dword-packed:
// A (16x32, MxK): lane = m + 16*ha ; half j: K = (j<8) ? 8*ha+j : 16+8*ha+(j-8)
//   -> adjacent even/odd K share a dword slot j>>1.
// B (32x16, KxN): lane = n + 16*hb ; half j: K = (j<8) ? 8*hb+j : 16+8*hb+(j-8)
//   -> adjacent even/odd K share a dword slot j>>1.
// ---------------------------------------------------------------------------

// ---------------------------------------------------------------------------
// Kernel 4: h = relu(x_gathered @ W1[e] + b1[e]), stored f16 to workspace.
// grid = (E, 16 Mtiles, H/128 Ntiles), block = 256 (8 waves, 16 N-cols each).
// Register double-buffered staging: global loads for tile i+1 overlap wmma i.
// ---------------------------------------------------------------------------
__global__ __launch_bounds__(256) void k_gemm1(
    const float* __restrict__ x, const float* __restrict__ W1,
    const float* __restrict__ b1,
    const int* __restrict__ counts, const int* __restrict__ slotRow,
    _Float16* __restrict__ hBuf)
{
    const int e = blockIdx.x, mtile = blockIdx.y, nt = blockIdx.z;
    const int cnt = counts[e];
    if (mtile * 16 >= cnt) return;
    const int slotBase = e * SLOTS_PER_E + mtile * 16;
    const int n0 = nt * 128;

    __shared__ __align__(32) unsigned int aBufU[16][2][8];   // 1 KB
    __shared__ __align__(32) unsigned int bBufU[128][2][8];  // 8 KB
    __shared__ int sRow[16];

    const int t = threadIdx.x;
    if (t < 16) sRow[t] = slotRow[slotBase + t];
    __syncthreads();

    const int wave = t >> 5, lane = t & 31;
    const int nl = lane & 15, hi = lane >> 4;

    // staging roles
    const int rA = t >> 4;            // A row 0..15
    const int cA = (t & 15) * 2;      // A col pair base (even)
    const int kB = (t >> 4) * 2;      // B k-row pair base (even)
    const int cB = (t & 15) * 8;      // B col base 0..120
    const int haA  = (cA >> 3) & 1;
    const int jA   = ((cA & 7) + (((cA >> 4) & 1) << 3)) >> 1;
    const int hbB  = (kB >> 3) & 1;
    const int sB   = ((kB & 7) + (((kB >> 4) & 1) << 3)) >> 1;
    const int rowA = sRow[rA];
    const float* We = W1 + (size_t)e * DD * HH;

    float f0 = 0.f, f1 = 0.f, w0[8], w1[8];
    {   // prologue: load K-tile 0 into registers
        if (rowA >= 0) {
            const float* xp = x + (size_t)rowA * DD + cA;
            f0 = xp[0]; f1 = xp[1];
        }
        const float* wp = We + (size_t)kB * HH + n0 + cB;
#pragma unroll
        for (int j = 0; j < 8; ++j) { w0[j] = wp[j]; w1[j] = wp[HH + j]; }
    }

    v8f c = {};
    for (int k0 = 0; k0 < DD; k0 += 32) {
        __syncthreads();
        // commit staged registers to LDS (dword-packed fragment layout)
        aBufU[rA][haA][jA] = pk2(f0, f1);
#pragma unroll
        for (int j = 0; j < 8; ++j)
            bBufU[cB + j][hbB][sB] = pk2(w0[j], w1[j]);
        __syncthreads();

        const int kn = k0 + 32;
        if (kn < DD) {   // prefetch next tile into registers (overlaps wmma)
            f0 = 0.f; f1 = 0.f;
            if (rowA >= 0) {
                const float* xp = x + (size_t)rowA * DD + kn + cA;
                f0 = xp[0]; f1 = xp[1];
            }
            const float* wp = We + (size_t)(kn + kB) * HH + n0 + cB;
#pragma unroll
            for (int j = 0; j < 8; ++j) { w0[j] = wp[j]; w1[j] = wp[HH + j]; }
            if (kn + 64 < DD)   // L2->WGP warm-up two tiles ahead
                __builtin_prefetch(We + (size_t)(kn + 64 + kB) * HH + n0 + cB, 0, 1);
        }

        const v16h a = *(const v16h*)&aBufU[nl][hi][0];
        const v16h b = *(const v16h*)&bBufU[wave * 16 + nl][hi][0];
        c = __builtin_amdgcn_wmma_f32_16x16x32_f16(false, a, false, b,
                                                   (short)0, c, false, false);
    }

    // epilogue: +b1, ReLU, f16 store. lane covers col n, rows m = v + 8*hi
    const int n = n0 + wave * 16 + nl;
    const float bias = b1[(size_t)e * HH + n];
#pragma unroll
    for (int v = 0; v < 8; ++v) {
        const int m = v + 8 * hi;
        float hv = c[v] + bias;
        hv = hv > 0.f ? hv : 0.f;
        hBuf[(size_t)(slotBase + m) * HH + n] = (_Float16)hv;
    }
}

// ---------------------------------------------------------------------------
// Kernel 5: out[row] += gate * (h @ W2[e] + b2[e])  (atomic f32, 2 adds/row)
// ---------------------------------------------------------------------------
__global__ __launch_bounds__(256) void k_gemm2(
    const _Float16* __restrict__ hBuf, const float* __restrict__ W2,
    const float* __restrict__ b2,
    const int* __restrict__ counts, const int* __restrict__ slotRow,
    const float* __restrict__ slotGate, float* __restrict__ out)
{
    const int e = blockIdx.x, mtile = blockIdx.y, nt = blockIdx.z;
    const int cnt = counts[e];
    if (mtile * 16 >= cnt) return;
    const int slotBase = e * SLOTS_PER_E + mtile * 16;
    const int n0 = nt * 128;

    __shared__ __align__(32) unsigned int aBufU[16][2][8];
    __shared__ __align__(32) unsigned int bBufU[128][2][8];
    __shared__ int   sRow[16];
    __shared__ float sGate[16];

    const int t = threadIdx.x;
    if (t < 16) { sRow[t] = slotRow[slotBase + t]; sGate[t] = slotGate[slotBase + t]; }
    __syncthreads();

    const int wave = t >> 5, lane = t & 31;
    const int nl = lane & 15, hi = lane >> 4;

    const int rA = t >> 4;
    const int cA = (t & 15) * 2;
    const int kB = (t >> 4) * 2;
    const int cB = (t & 15) * 8;
    const int haA = (cA >> 3) & 1;
    const int jA  = ((cA & 7) + (((cA >> 4) & 1) << 3)) >> 1;
    const int hbB = (kB >> 3) & 1;
    const int sB  = ((kB & 7) + (((kB >> 4) & 1) << 3)) >> 1;
    const float* We = W2 + (size_t)e * HH * HH;
    const _Float16* hRow = hBuf + (size_t)(slotBase + rA) * HH;

    unsigned int av; float w0[8], w1[8];
    {   // prologue: K-tile 0
        av = *(const unsigned int*)(hRow + cA);   // two f16, already fragment-packed
        const float* wp = We + (size_t)kB * HH + n0 + cB;
#pragma unroll
        for (int j = 0; j < 8; ++j) { w0[j] = wp[j]; w1[j] = wp[HH + j]; }
    }

    v8f c = {};
    for (int k0 = 0; k0 < HH; k0 += 32) {
        __syncthreads();
        aBufU[rA][haA][jA] = av;
#pragma unroll
        for (int j = 0; j < 8; ++j)
            bBufU[cB + j][hbB][sB] = pk2(w0[j], w1[j]);
        __syncthreads();

        const int kn = k0 + 32;
        if (kn < HH) {
            av = *(const unsigned int*)(hRow + kn + cA);
            const float* wp = We + (size_t)(kn + kB) * HH + n0 + cB;
#pragma unroll
            for (int j = 0; j < 8; ++j) { w0[j] = wp[j]; w1[j] = wp[HH + j]; }
            if (kn + 64 < HH)
                __builtin_prefetch(We + (size_t)(kn + 64 + kB) * HH + n0 + cB, 0, 1);
        }

        const v16h a = *(const v16h*)&aBufU[nl][hi][0];
        const v16h b = *(const v16h*)&bBufU[wave * 16 + nl][hi][0];
        c = __builtin_amdgcn_wmma_f32_16x16x32_f16(false, a, false, b,
                                                   (short)0, c, false, false);
    }

    const int n = n0 + wave * 16 + nl;
    const float bias = b2[(size_t)e * HH + n];
#pragma unroll
    for (int v = 0; v < 8; ++v) {
        const int m = v + 8 * hi;
        const int row = sRow[m];
        if (row >= 0)
            atomicAdd(&out[(size_t)row * HH + n], (c[v] + bias) * sGate[m]);
    }
}

// ---------------------------------------------------------------------------
extern "C" void kernel_launch(void* const* d_in, const int* in_sizes, int n_in,
                              void* d_out, int out_size, void* d_ws, size_t ws_size,
                              hipStream_t stream) {
    (void)in_sizes; (void)n_in; (void)out_size; (void)ws_size;
    const float* x    = (const float*)d_in[0];
    const float* W1   = (const float*)d_in[1];
    const float* b1   = (const float*)d_in[2];
    const float* W2   = (const float*)d_in[3];
    const float* b2   = (const float*)d_in[4];
    const float* gW   = (const float*)d_in[5];
    const float* gB   = (const float*)d_in[6];
    const float* lgW  = (const float*)d_in[7];
    const float* lgB  = (const float*)d_in[8];

    float* out     = (float*)d_out;
    float* gateOut = out + (size_t)TOKENS * HH;   // tuple: (out, gate_scores)

    char* ws = (char*)d_ws;
    float*     localScores = (float*)(ws);                                   // 64 KB
    int*       counts      = (int*)  (ws + 65536);                           // 32 B (pad 256)
    int*       slotRow     = (int*)  (ws + 65536 + 256);                     // 8 KB
    float*     slotGate    = (float*)(ws + 65536 + 256 + 8192);              // 8 KB
    _Float16*  hBuf        = (_Float16*)(ws + 65536 + 256 + 8192 + 8192);    // 4 MB

    k_gate<<<TOKENS / 8, 256, 0, stream>>>(x, gW, gB, lgW, lgB, gateOut, localScores);
    k_dispatch<<<1, 256, 0, stream>>>(localScores, gateOut, counts, slotRow, slotGate);
    k_zero<<<(TOKENS * (HH / 4)) / 256, 256, 0, stream>>>((v4f*)out);
    k_gemm1<<<dim3(EE, SLOTS_PER_E / 16, HH / 128), 256, 0, stream>>>(
        x, W1, b1, counts, slotRow, hBuf);
    k_gemm2<<<dim3(EE, SLOTS_PER_E / 16, HH / 128), 256, 0, stream>>>(
        hBuf, W2, b2, counts, slotRow, slotGate, out);
}
